// PhysicsInformedAttention_83184926589116
// MI455X (gfx1250) — matrix operand order repaired and checked by reading, not compiled, not run on hardware
//
#include <hip/hip_runtime.h>

#define SEQ   2048
#define FDIM  1024
#define FIN   1536
#define HEADS 16
#define BATCH 4
#define MROWS (BATCH * SEQ)   // 8192

typedef __attribute__((ext_vector_type(16))) __bf16 v16bf;
typedef __attribute__((ext_vector_type(8)))  float  v8f;

union Frag16 {
    uint4  u[2];
    __bf16 e[16];
    v16bf  v;
};

// ---------------------------------------------------------------------------
// Pack a row-major f32 weight [K,N] into bf16 B-fragment order:
// packed[((nt*(K/32)+kt)*32 + lane)*16 + i] = W[kt*32 + (i>>3)*16 + (lane>>4)*8 + (i&7)][nt*16 + (lane&15)]
// so each lane's 16 values for one 32x16 tile are contiguous (two b128 loads).
// ---------------------------------------------------------------------------
__global__ __launch_bounds__(256) void pack_w_kernel(const float* __restrict__ W,
                                                     __bf16* __restrict__ out,
                                                     int K, int N) {
    size_t idx = (size_t)blockIdx.x * blockDim.x + threadIdx.x;
    if (idx >= (size_t)K * N) return;
    int i    = (int)(idx & 15);
    int lane = (int)((idx >> 4) & 31);
    size_t t = idx >> 9;              // tile index = nt*(K/32) + kt
    int KT = K >> 5;
    int kt = (int)(t % KT);
    int nt = (int)(t / KT);
    int k = kt * 32 + (i >> 3) * 16 + (lane >> 4) * 8 + (i & 7);
    int n = nt * 16 + (lane & 15);
    out[idx] = (__bf16)W[(size_t)k * N + n];
}

// ---------------------------------------------------------------------------
// ftab[d] = distance_scale*log1p(d) - d*exp(-localization_length)
// ---------------------------------------------------------------------------
__global__ __launch_bounds__(256) void ftab_kernel(const float* __restrict__ dsc,
                                                   const float* __restrict__ lln,
                                                   float* __restrict__ ftab, int n) {
    int i = blockIdx.x * blockDim.x + threadIdx.x;
    if (i < n) {
        float d = (float)i;
        ftab[i] = dsc[0] * log1pf(d) - d * __expf(-lln[0]);
    }
}

__device__ inline void load_cvt_frag(const float* p, Frag16& a) {
    float4 x0 = *(const float4*)(p);
    float4 x1 = *(const float4*)(p + 4);
    float4 y0 = *(const float4*)(p + 16);
    float4 y1 = *(const float4*)(p + 20);
    a.e[0]=(__bf16)x0.x;  a.e[1]=(__bf16)x0.y;  a.e[2]=(__bf16)x0.z;  a.e[3]=(__bf16)x0.w;
    a.e[4]=(__bf16)x1.x;  a.e[5]=(__bf16)x1.y;  a.e[6]=(__bf16)x1.z;  a.e[7]=(__bf16)x1.w;
    a.e[8]=(__bf16)y0.x;  a.e[9]=(__bf16)y0.y;  a.e[10]=(__bf16)y0.z; a.e[11]=(__bf16)y0.w;
    a.e[12]=(__bf16)y1.x; a.e[13]=(__bf16)y1.y; a.e[14]=(__bf16)y1.z; a.e[15]=(__bf16)y1.w;
}

// ---------------------------------------------------------------------------
// Projection GEMM: out = cat(act, sym) @ W + bias, bf16 out.
// Wave computes a 32x64 tile (2 M-tiles x 4 N-tiles, 8 WMMA per 32-k step).
// k-loop split: kt<32 -> f32 activations; kt>=32 -> (s&1)*Wsym (A frag shared
// by both M-tiles since rows m and m+16 have equal parity).
// mode 0: row-major bf16 [M, F].  mode 1: V transposed [B, H, 64, S] bf16.
// ---------------------------------------------------------------------------
__global__ __launch_bounds__(256) void proj_gemm_kernel(
        const float* __restrict__ act, const float* __restrict__ wsym,
        const __bf16* __restrict__ Wp, const float* __restrict__ bias,
        __bf16* __restrict__ out, int K, int mode) {
    const int lane   = threadIdx.x & 31;
    const int wave   = threadIdx.x >> 5;
    const int lane15 = lane & 15;
    const int half   = lane >> 4;
    const int m0  = (blockIdx.x * 8 + wave) * 32;
    const int n0g = blockIdx.y * 4;           // first 16-col tile
    const int KT  = K >> 5;                   // 48
    const int KT_ACT = FDIM >> 5;             // 32

    const float* arow0 = act + (size_t)(m0 + lane15) * FDIM;
    const float* arow1 = act + (size_t)(m0 + 16 + lane15) * FDIM;
    const float par = ((m0 + lane15) & 1) ? 1.0f : 0.0f;

    v8f acc[2][4];
    v8f zero = {};
    #pragma unroll
    for (int mt = 0; mt < 2; ++mt)
        #pragma unroll
        for (int nt = 0; nt < 4; ++nt) acc[mt][nt] = zero;

    const size_t bstride = (size_t)KT * 512;  // next n-tile in packed W

    // ---- activation region ----
    for (int kt = 0; kt < KT_ACT; ++kt) {
        const int k1 = (kt << 5) + half * 8;
        const __bf16* bbase = Wp + ((size_t)(n0g * KT + kt) * 32 + lane) * 16;
        __builtin_prefetch(bbase + 512, 0, 3);     // next kt tile
        v16bf bfr[4];
        #pragma unroll
        for (int nt = 0; nt < 4; ++nt)
            bfr[nt] = *(const v16bf*)(bbase + (size_t)nt * bstride);
        Frag16 a0, a1;
        load_cvt_frag(arow0 + k1, a0);
        load_cvt_frag(arow1 + k1, a1);
        #pragma unroll
        for (int nt = 0; nt < 4; ++nt) {
            acc[0][nt] = __builtin_amdgcn_wmma_f32_16x16x32_bf16(
                false, a0.v, false, bfr[nt], (short)0, acc[0][nt], false, false);
            acc[1][nt] = __builtin_amdgcn_wmma_f32_16x16x32_bf16(
                false, a1.v, false, bfr[nt], (short)0, acc[1][nt], false, false);
        }
    }
    // ---- spin-symmetry region (shared A fragment) ----
    for (int kt = KT_ACT; kt < KT; ++kt) {
        const int k1 = (kt << 5) + half * 8 - FDIM;
        const __bf16* bbase = Wp + ((size_t)(n0g * KT + kt) * 32 + lane) * 16;
        __builtin_prefetch(bbase + 512, 0, 3);
        v16bf bfr[4];
        #pragma unroll
        for (int nt = 0; nt < 4; ++nt)
            bfr[nt] = *(const v16bf*)(bbase + (size_t)nt * bstride);
        Frag16 a;
        #pragma unroll
        for (int i = 0; i < 8; ++i) a.e[i] = (__bf16)(wsym[k1 + i] * par);
        #pragma unroll
        for (int i = 0; i < 8; ++i) a.e[8 + i] = (__bf16)(wsym[k1 + 16 + i] * par);
        #pragma unroll
        for (int nt = 0; nt < 4; ++nt) {
            acc[0][nt] = __builtin_amdgcn_wmma_f32_16x16x32_bf16(
                false, a.v, false, bfr[nt], (short)0, acc[0][nt], false, false);
            acc[1][nt] = __builtin_amdgcn_wmma_f32_16x16x32_bf16(
                false, a.v, false, bfr[nt], (short)0, acc[1][nt], false, false);
        }
    }

    #pragma unroll
    for (int mt = 0; mt < 2; ++mt) {
        #pragma unroll
        for (int nt = 0; nt < 4; ++nt) {
            #pragma unroll
            for (int r = 0; r < 8; ++r) {
                const int mrow = m0 + mt * 16 + r + half * 8;
                const int ncol = (n0g + nt) * 16 + lane15;
                const float vv = acc[mt][nt][r] + bias[ncol];
                if (mode == 0) {
                    out[(size_t)mrow * FDIM + ncol] = (__bf16)vv;
                } else {
                    const int hh = ncol >> 6, dd = ncol & 63;
                    const int bb = mrow >> 11, ss = mrow & (SEQ - 1);
                    out[(((size_t)(bb * HEADS + hh) * 64 + dd) << 11) + ss] = (__bf16)vv;
                }
            }
        }
    }
}

// ---------------------------------------------------------------------------
// Flash attention. Grid (B*H, S/128); block = 128 threads = 4 waves.
// Wave owns 32 q-rows (2 tiles); per 32-key step the 4 K-fragments and 4
// V-fragments feed 8 QK^T WMMAs and 8 PV WMMAs. Distance-bias table in LDS.
// ---------------------------------------------------------------------------
__global__ __launch_bounds__(128) void attn_kernel(
        const __bf16* __restrict__ Qb, const __bf16* __restrict__ Kb,
        const __bf16* __restrict__ VT, const unsigned char* __restrict__ mask,
        const float* __restrict__ ftab, const float* __restrict__ psp,
        __bf16* __restrict__ ATT) {
    __shared__ __bf16 plds[4][2][16 * 32];
    __shared__ float  flds[SEQ];

    const int lane   = threadIdx.x & 31;
    const int wave   = threadIdx.x >> 5;
    const int lane15 = lane & 15;
    const int half   = lane >> 4;
    const int b = blockIdx.x >> 4;
    const int h = blockIdx.x & 15;
    const int q0 = blockIdx.y * 128 + wave * 32;
    const float ps = psp[0];

    for (int i = threadIdx.x; i < SEQ; i += 128) flds[i] = ftab[i];
    __syncthreads();

    // Q fragments: 2 M-tiles x 2 d-steps, resident for whole loop
    Frag16 qa[2][2];
    #pragma unroll
    for (int mt = 0; mt < 2; ++mt) {
        const __bf16* qrow = Qb + ((size_t)(b * SEQ + q0 + mt * 16 + lane15) * FDIM) + h * 64;
        #pragma unroll
        for (int t = 0; t < 2; ++t) {
            qa[mt][t].u[0] = *(const uint4*)(qrow + t * 32 + half * 16);
            qa[mt][t].u[1] = *(const uint4*)(qrow + t * 32 + half * 16 + 8);
        }
    }

    float mi[2][8], li[2][8];
    v8f zero = {};
    v8f oacc[2][4];
    #pragma unroll
    for (int mt = 0; mt < 2; ++mt) {
        #pragma unroll
        for (int r = 0; r < 8; ++r) { mi[mt][r] = -3.4028235e38f; li[mt][r] = 0.0f; }
        #pragma unroll
        for (int nt = 0; nt < 4; ++nt) oacc[mt][nt] = zero;
    }

    const unsigned char* mrowb = mask + (size_t)b * SEQ * SEQ;

    for (int kb0 = 0; kb0 < SEQ; kb0 += 32) {
        // ---- K fragments (4), then 8 WMMA for S = Q.K^T ----
        v16bf kf[2][2];
        #pragma unroll
        for (int t = 0; t < 2; ++t) {
            const int kcol = kb0 + t * 16 + lane15;
            const __bf16* krow = Kb + ((size_t)(b * SEQ + kcol) * FDIM) + h * 64 + half * 16;
            kf[t][0] = *(const v16bf*)(krow);
            kf[t][1] = *(const v16bf*)(krow + 32);
        }
        v8f st[2][2];
        #pragma unroll
        for (int mt = 0; mt < 2; ++mt)
            #pragma unroll
            for (int t = 0; t < 2; ++t) {
                v8f c = zero;
                c = __builtin_amdgcn_wmma_f32_16x16x32_bf16(false, qa[mt][0].v, false, kf[t][0], (short)0, c, false, false);
                c = __builtin_amdgcn_wmma_f32_16x16x32_bf16(false, qa[mt][1].v, false, kf[t][1], (short)0, c, false, false);
                st[mt][t] = c;
            }
        // ---- fused bias + parity + mask + scale; online softmax per M-tile ----
        #pragma unroll
        for (int mt = 0; mt < 2; ++mt) {
            const int iBase = q0 + mt * 16 + half * 8;
            float sc[2][8];
            #pragma unroll
            for (int t = 0; t < 2; ++t) {
                const int j = kb0 + t * 16 + lane15;
                #pragma unroll
                for (int r = 0; r < 8; ++r) {
                    const int i = iBase + r;
                    const int dist = (i > j) ? (i - j) : (j - i);
                    const float bias = flds[dist] + ps * (1.0f - 2.0f * (float)((i + j) & 1));
                    float v = (st[mt][t][r] + bias) * 0.125f;
                    if (!mrowb[(size_t)i * SEQ + j]) v = -3.4028235e38f;
                    sc[t][r] = v;
                }
            }
            #pragma unroll
            for (int r = 0; r < 8; ++r) {
                float v = fmaxf(sc[0][r], sc[1][r]);
                v = fmaxf(v, __shfl_xor(v, 1, 32));
                v = fmaxf(v, __shfl_xor(v, 2, 32));
                v = fmaxf(v, __shfl_xor(v, 4, 32));
                v = fmaxf(v, __shfl_xor(v, 8, 32));
                const float mn   = fmaxf(mi[mt][r], v);
                const float corr = __expf(mi[mt][r] - mn);
                mi[mt][r] = mn;
                const float p0 = __expf(sc[0][r] - mn);
                const float p1 = __expf(sc[1][r] - mn);
                float rs = p0 + p1;
                rs += __shfl_xor(rs, 1, 32);
                rs += __shfl_xor(rs, 2, 32);
                rs += __shfl_xor(rs, 4, 32);
                rs += __shfl_xor(rs, 8, 32);
                li[mt][r] = li[mt][r] * corr + rs;
                #pragma unroll
                for (int nt = 0; nt < 4; ++nt) oacc[mt][nt][r] *= corr;
                const int rloc = r + half * 8;
                plds[wave][mt][rloc * 32 + lane15]      = (__bf16)p0;
                plds[wave][mt][rloc * 32 + 16 + lane15] = (__bf16)p1;
            }
        }
        asm volatile("s_wait_dscnt 0" ::: "memory");   // LDS RAW within wave
        // ---- P in A-fragment layout, V fragments (4), 8 WMMA for O += P.V ----
        Frag16 pf[2];
        #pragma unroll
        for (int mt = 0; mt < 2; ++mt) {
            const __bf16* pl = &plds[wave][mt][0] + lane15 * 32;
            pf[mt].u[0] = *(const uint4*)(pl + half * 16);
            pf[mt].u[1] = *(const uint4*)(pl + half * 16 + 8);
        }
        v16bf vf[4];
        #pragma unroll
        for (int nt = 0; nt < 4; ++nt) {
            const __bf16* vrow = VT + (((size_t)(b * HEADS + h) * 64 + nt * 16 + lane15) << 11)
                               + kb0 + half * 16;
            vf[nt] = *(const v16bf*)vrow;
        }
        #pragma unroll
        for (int mt = 0; mt < 2; ++mt)
            #pragma unroll
            for (int nt = 0; nt < 4; ++nt)
                oacc[mt][nt] = __builtin_amdgcn_wmma_f32_16x16x32_bf16(
                    false, pf[mt].v, false, vf[nt], (short)0, oacc[mt][nt], false, false);
    }
    // ---- normalize and emit bf16 attn [B,S,F] ----
    #pragma unroll
    for (int mt = 0; mt < 2; ++mt) {
        #pragma unroll
        for (int nt = 0; nt < 4; ++nt) {
            #pragma unroll
            for (int r = 0; r < 8; ++r) {
                const int i = q0 + mt * 16 + r + half * 8;
                const float den = li[mt][r];
                const float vv = (den > 0.0f) ? oacc[mt][nt][r] / den : 0.0f;
                ATT[((size_t)(b * SEQ + i)) * FDIM + h * 64 + nt * 16 + lane15] = (__bf16)vv;
            }
        }
    }
}

// ---------------------------------------------------------------------------
// Output GEMM: d_out[M,1024] = attn(bf16) @ Wo + bo, f32 out. 32x64 per wave.
// ---------------------------------------------------------------------------
__global__ __launch_bounds__(256) void out_gemm_kernel(
        const __bf16* __restrict__ A, const __bf16* __restrict__ Wp,
        const float* __restrict__ bias, float* __restrict__ out) {
    const int lane   = threadIdx.x & 31;
    const int wave   = threadIdx.x >> 5;
    const int lane15 = lane & 15;
    const int half   = lane >> 4;
    const int m0  = (blockIdx.x * 8 + wave) * 32;
    const int n0g = blockIdx.y * 4;
    const int KT  = FDIM >> 5;                 // 32

    const __bf16* arow0 = A + (size_t)(m0 + lane15) * FDIM;
    const __bf16* arow1 = A + (size_t)(m0 + 16 + lane15) * FDIM;

    v8f acc[2][4];
    v8f zero = {};
    #pragma unroll
    for (int mt = 0; mt < 2; ++mt)
        #pragma unroll
        for (int nt = 0; nt < 4; ++nt) acc[mt][nt] = zero;

    const size_t bstride = (size_t)KT * 512;

    for (int kt = 0; kt < KT; ++kt) {
        const int kb = (kt << 5) + half * 8;
        const __bf16* bbase = Wp + ((size_t)(n0g * KT + kt) * 32 + lane) * 16;
        __builtin_prefetch(bbase + 512, 0, 3);
        v16bf bfr[4];
        #pragma unroll
        for (int nt = 0; nt < 4; ++nt)
            bfr[nt] = *(const v16bf*)(bbase + (size_t)nt * bstride);
        Frag16 a0, a1;
        a0.u[0] = *(const uint4*)(arow0 + kb);
        a0.u[1] = *(const uint4*)(arow0 + kb + 16);
        a1.u[0] = *(const uint4*)(arow1 + kb);
        a1.u[1] = *(const uint4*)(arow1 + kb + 16);
        #pragma unroll
        for (int nt = 0; nt < 4; ++nt) {
            acc[0][nt] = __builtin_amdgcn_wmma_f32_16x16x32_bf16(
                false, a0.v, false, bfr[nt], (short)0, acc[0][nt], false, false);
            acc[1][nt] = __builtin_amdgcn_wmma_f32_16x16x32_bf16(
                false, a1.v, false, bfr[nt], (short)0, acc[1][nt], false, false);
        }
    }
    #pragma unroll
    for (int mt = 0; mt < 2; ++mt)
        #pragma unroll
        for (int nt = 0; nt < 4; ++nt)
            #pragma unroll
            for (int r = 0; r < 8; ++r) {
                const int mrow = m0 + mt * 16 + r + half * 8;
                const int ncol = (n0g + nt) * 16 + lane15;
                out[(size_t)mrow * FDIM + ncol] = acc[mt][nt][r] + bias[ncol];
            }
}

// ---------------------------------------------------------------------------
extern "C" void kernel_launch(void* const* d_in, const int* in_sizes, int n_in,
                              void* d_out, int out_size, void* d_ws, size_t ws_size,
                              hipStream_t stream) {
    const float* kv  = (const float*)d_in[0];
    const float* q   = (const float*)d_in[1];
    const unsigned char* mask = (const unsigned char*)d_in[2];
    const float* Wsym = (const float*)d_in[3];
    const float* Wq  = (const float*)d_in[4];
    const float* bq  = (const float*)d_in[5];
    const float* Wk  = (const float*)d_in[6];
    const float* bk  = (const float*)d_in[7];
    const float* Wv  = (const float*)d_in[8];
    const float* bv  = (const float*)d_in[9];
    const float* Wo  = (const float*)d_in[10];
    const float* bo  = (const float*)d_in[11];
    const float* dsc = (const float*)d_in[12];
    const float* pst = (const float*)d_in[13];
    const float* lln = (const float*)d_in[14];
    float* out = (float*)d_out;

    char* ws = (char*)d_ws;
    size_t off = 0;
    auto take = [&](size_t bytes) -> char* {
        char* p = ws + off;
        off += (bytes + 255) & ~(size_t)255;
        return p;
    };
    __bf16* WqP = (__bf16*)take((size_t)FIN * FDIM * 2);
    __bf16* WkP = (__bf16*)take((size_t)FIN * FDIM * 2);
    __bf16* WvP = (__bf16*)take((size_t)FIN * FDIM * 2);
    __bf16* WoP = (__bf16*)take((size_t)FDIM * FDIM * 2);
    __bf16* Qb  = (__bf16*)take((size_t)MROWS * FDIM * 2);
    __bf16* Kb  = (__bf16*)take((size_t)MROWS * FDIM * 2);
    __bf16* VT  = (__bf16*)take((size_t)MROWS * FDIM * 2);
    __bf16* ATT = (__bf16*)take((size_t)MROWS * FDIM * 2);
    float*  FTB = (float*)take((size_t)SEQ * 4);

    // 1) weight packing + bias table
    pack_w_kernel<<<(FIN * FDIM) / 256, 256, 0, stream>>>(Wq, WqP, FIN, FDIM);
    pack_w_kernel<<<(FIN * FDIM) / 256, 256, 0, stream>>>(Wk, WkP, FIN, FDIM);
    pack_w_kernel<<<(FIN * FDIM) / 256, 256, 0, stream>>>(Wv, WvP, FIN, FDIM);
    pack_w_kernel<<<(FDIM * FDIM) / 256, 256, 0, stream>>>(Wo, WoP, FDIM, FDIM);
    ftab_kernel<<<SEQ / 256, 256, 0, stream>>>(dsc, lln, FTB, SEQ);

    // 2) Q/K/V projections (V written transposed per head for P.V fragments)
    dim3 g1(MROWS / 256, FDIM / 64);
    proj_gemm_kernel<<<g1, 256, 0, stream>>>(q,  Wsym, WqP, bq, Qb, FIN, 0);
    proj_gemm_kernel<<<g1, 256, 0, stream>>>(kv, Wsym, WkP, bk, Kb, FIN, 0);
    proj_gemm_kernel<<<g1, 256, 0, stream>>>(kv, Wsym, WvP, bv, VT, FIN, 1);

    // 3) flash attention with fused physics biases
    attn_kernel<<<dim3(BATCH * HEADS, SEQ / 128), 128, 0, stream>>>(
        Qb, Kb, VT, mask, FTB, pst, ATT);

    // 4) output projection (f32 out + bo)
    out_gemm_kernel<<<g1, 256, 0, stream>>>(ATT, WoP, bo, out);
}
